// EdgePredictorExplicit_5068061409528
// MI455X (gfx1250) — matrix operand order repaired and checked by reading, not compiled, not run on hardware
//
#include <hip/hip_runtime.h>
#include <hip/hip_bf16.h>

typedef __attribute__((ext_vector_type(16))) _Float16 v16h;
typedef __attribute__((ext_vector_type(8)))  _Float16 v8h;
typedef __attribute__((ext_vector_type(8)))  float    v8f;

#define B_DIM 4
#define V_DIM 1024
#define D_DIM 3
#define H_DIM 128
#define E_DIM ((V_DIM * (V_DIM - 1)) / 2)   /* 523776 */
#define ROWS  (B_DIM * E_DIM)               /* 2095104 */

#define WAVES_PER_BLOCK 4
#define BLOCK_THREADS (32 * WAVES_PER_BLOCK)
#define NTILES (ROWS / 16)                   /* 130944, exact */
#define GRID_BLOCKS 4096

// Invert e -> (i, j), i < j, triu row-major order: e = i*(2V-1-i)/2 + (j-i-1)
__device__ __forceinline__ void e_to_ij(int e, int& io, int& jo) {
    const float a = 2.0f * V_DIM - 1.0f;  // 2047
    float disc = a * a - 8.0f * (float)e;
    disc = disc > 0.0f ? disc : 0.0f;
    int i = (int)((a - __builtin_sqrtf(disc)) * 0.5f);
    if (i < 0) i = 0;
    if (i > V_DIM - 2) i = V_DIM - 2;
    while (i > 0 && (i * (2 * V_DIM - 1 - i)) / 2 > e) --i;
    while (i < V_DIM - 2 && ((i + 1) * (2 * V_DIM - 1 - (i + 1))) / 2 <= e) ++i;
    io = i;
    jo = e - (i * (2 * V_DIM - 1 - i)) / 2 + i + 1;
}

__global__ __launch_bounds__(BLOCK_THREADS)
void edge_mlp_kernel(const float* __restrict__ vtx,
                     const float* __restrict__ W1, const float* __restrict__ b1,
                     const float* __restrict__ W2, const float* __restrict__ b2,
                     const float* __restrict__ W3, const float* __restrict__ b3,
                     float* __restrict__ out)
{
    // W1 transposed and zero-padded to full K=32 so every lane's B-fragment load is
    // one unconditional aligned v16h read (lanes 16-31 / K>=6 read stored zeros).
    __shared__ __attribute__((aligned(32))) _Float16 sW1T[H_DIM * 32];           // [col][k0..31]
    __shared__ __attribute__((aligned(32))) _Float16 sW2T[(H_DIM / 2) * H_DIM];  // [col][k0..127] (init only)
    __shared__ float sW3[H_DIM / 2];
    __shared__ float sB1[H_DIM];
    __shared__ float sB2[H_DIM / 2];
    // per-wave staging: one 16 x 128 f16 hidden tile
    __shared__ __attribute__((aligned(32))) _Float16 sH1[WAVES_PER_BLOCK][16 * H_DIM];

    const int tid = threadIdx.x;

    // ---- stage weights into LDS (f16), one uniform barrier pair ----
    for (int t = tid; t < H_DIM * 32; t += BLOCK_THREADS) sW1T[t] = (_Float16)0.0f;
    __syncthreads();
    for (int t = tid; t < 2 * D_DIM * H_DIM; t += BLOCK_THREADS) {
        int k = t / H_DIM, c = t % H_DIM;
        sW1T[c * 32 + k] = (_Float16)W1[t];
    }
    for (int t = tid; t < H_DIM * (H_DIM / 2); t += BLOCK_THREADS) {
        int k = t / (H_DIM / 2), c = t % (H_DIM / 2);
        sW2T[c * H_DIM + k] = (_Float16)W2[t];
    }
    for (int t = tid; t < H_DIM / 2; t += BLOCK_THREADS) { sW3[t] = W3[t]; sB2[t] = b2[t]; }
    for (int t = tid; t < H_DIM; t += BLOCK_THREADS) sB1[t] = b1[t];
    __syncthreads();

    const int wave = tid >> 5;
    const int lane = tid & 31;
    const int half = lane >> 4;   // 0: lanes 0-15, 1: lanes 16-31
    const int l15  = lane & 15;

    // ---- W2 B-fragments: fully register-resident, loop-invariant (16 x v16h = 128 VGPRs) ----
    // Fragment (kt, n2): col = 16*n2 + l15; K = 32*kt + 16*half + t (contiguous in sW2T)
    v16h bW2[4][4];
    #pragma unroll
    for (int kt = 0; kt < 4; ++kt)
        #pragma unroll
        for (int n2 = 0; n2 < 4; ++n2)
            bW2[kt][n2] = *(const v16h*)&sW2T[(n2 * 16 + l15) * H_DIM + kt * 32 + 16 * half];

    float bias1[8];
    #pragma unroll
    for (int n = 0; n < H_DIM / 16; ++n) bias1[n] = sB1[n * 16 + l15];
    float w3v[4], bias2v[4];
    #pragma unroll
    for (int n2 = 0; n2 < 4; ++n2) {
        w3v[n2]    = sW3[n2 * 16 + l15];
        bias2v[n2] = sB2[n2 * 16 + l15];
    }
    const float bias3 = b3[0];

    _Float16* const myH1 = &sH1[wave][0];
    // Base of this lane's L1 B-fragment column strip (col = 16n + l15 -> +n*16*32 halfs)
    const _Float16* const myW1 = &sW1T[l15 * 32 + 16 * half];

    const int gwave  = blockIdx.x * WAVES_PER_BLOCK + wave;
    const int stride = GRID_BLOCKS * WAVES_PER_BLOCK;

    // ---- persistent grid-stride loop: one 16-row tile per iteration ----
    for (int tile = gwave; tile < NTILES; tile += stride) {
        const int rowBase = tile * 16;

        // ===== Layer 1 A-fragment: 16 rows x K6 (zero-padded to 32); lanes 0-15 only =====
        v16h a1;
        for (int t = 0; t < 16; ++t) a1[t] = (_Float16)0.0f;
        if (half == 0) {
            int row = rowBase + l15;
            if (row >= ROWS) row = ROWS - 1;
            int b = row / E_DIM;
            int e = row - b * E_DIM;
            int i, j;
            e_to_ij(e, i, j);
            const float* pi = vtx + ((long)b * V_DIM + i) * D_DIM;
            const float* pj = vtx + ((long)b * V_DIM + j) * D_DIM;
            a1[0] = (_Float16)pi[0]; a1[1] = (_Float16)pi[1]; a1[2] = (_Float16)pi[2];
            a1[3] = (_Float16)pj[0]; a1[4] = (_Float16)pj[1]; a1[5] = (_Float16)pj[2];
        }

        // ===== Layer 1: h1 = relu(feat @ W1 + b1) -> LDS, software-pipelined =====
        // Rolling 2-deep buffer: issue WMMA(n+1) before consuming c(n) so the
        // WMMA->VALU RAW hazard slots fill with the previous tile's epilogue.
        v8f c1buf[2];
        c1buf[0] = __builtin_amdgcn_wmma_f32_16x16x32_f16(
            false, a1, false, *(const v16h*)&myW1[0], (short)0, (v8f){}, false, false);
        #pragma unroll
        for (int n = 0; n < H_DIM / 16; ++n) {
            if (n + 1 < H_DIM / 16) {
                const v16h bf = *(const v16h*)&myW1[(n + 1) * 16 * 32];
                c1buf[(n + 1) & 1] = __builtin_amdgcn_wmma_f32_16x16x32_f16(
                    false, a1, false, bf, (short)0, (v8f){}, false, false);
            }
            const v8f c = c1buf[n & 1];
            #pragma unroll
            for (int v = 0; v < 8; ++v) {
                float x = c[v] + bias1[n];
                x = x > 0.0f ? x : 0.0f;
                // C layout: row = v + 8*half, col = 16n + l15
                myH1[(v + 8 * half) * H_DIM + (n * 16 + l15)] = (_Float16)x;
            }
        }
        // Per-wave private LDS region; DS ops are in-order within a wave -> no barrier.

        // ===== Layer 2: 16x64 tile, K=128; B entirely from registers =====
        v8f c2[4];
        #pragma unroll
        for (int n2 = 0; n2 < 4; ++n2)
            #pragma unroll
            for (int v = 0; v < 8; ++v) c2[n2][v] = 0.0f;

        #pragma unroll
        for (int kt = 0; kt < 4; ++kt) {
            // A-fragment from LDS: row = l15; K = 32kt + 16*half + t (contiguous)
            const int kbase = kt * 32 + 16 * half;
            const v8h lo = *(const v8h*)&myH1[l15 * H_DIM + kbase];
            const v8h hi = *(const v8h*)&myH1[l15 * H_DIM + kbase + 8];
            v16h a2;
            #pragma unroll
            for (int t = 0; t < 8; ++t) { a2[t] = lo[t]; a2[t + 8] = hi[t]; }
            #pragma unroll
            for (int n2 = 0; n2 < 4; ++n2)
                c2[n2] = __builtin_amdgcn_wmma_f32_16x16x32_f16(false, a2, false, bW2[kt][n2],
                                                               (short)0, c2[n2], false, false);
        }

        // ===== Layer 3 (64 -> 1) + sigmoid, reduced straight from C layout =====
        float partial[8];
        #pragma unroll
        for (int v = 0; v < 8; ++v) partial[v] = 0.0f;
        #pragma unroll
        for (int n2 = 0; n2 < 4; ++n2) {
            #pragma unroll
            for (int v = 0; v < 8; ++v) {
                float x = c2[n2][v] + bias2v[n2];
                x = x > 0.0f ? x : 0.0f;
                partial[v] += x * w3v[n2];
            }
        }
        // sum over the 16-lane column group (lanes 0-15 and 16-31 reduce independently)
        #pragma unroll
        for (int off = 1; off < 16; off <<= 1)
            #pragma unroll
            for (int v = 0; v < 8; ++v)
                partial[v] += __shfl_xor(partial[v], off, 32);

        if (l15 == 0) {
            #pragma unroll
            for (int v = 0; v < 8; ++v) {
                int row = rowBase + v + 8 * half;
                if (row < ROWS) {
                    float x = partial[v] + bias3;
                    out[row] = 1.0f / (1.0f + __expf(-x));
                }
            }
        }
    }
}

// edge_indices [E, 2] appended after the B*E probabilities (stored as float per d_out dtype)
__global__ void edge_idx_kernel(float* __restrict__ out_idx)
{
    int e = blockIdx.x * blockDim.x + threadIdx.x;
    if (e >= E_DIM) return;
    int i, j;
    e_to_ij(e, i, j);
    out_idx[2 * e]     = (float)i;
    out_idx[2 * e + 1] = (float)j;
}

extern "C" void kernel_launch(void* const* d_in, const int* in_sizes, int n_in,
                              void* d_out, int out_size, void* d_ws, size_t ws_size,
                              hipStream_t stream) {
    const float* vtx = (const float*)d_in[0];
    const float* W1  = (const float*)d_in[1];
    const float* b1  = (const float*)d_in[2];
    const float* W2  = (const float*)d_in[3];
    const float* b2  = (const float*)d_in[4];
    const float* W3  = (const float*)d_in[5];
    const float* b3  = (const float*)d_in[6];
    float* out = (float*)d_out;

    edge_mlp_kernel<<<GRID_BLOCKS, BLOCK_THREADS, 0, stream>>>(vtx, W1, b1, W2, b2, W3, b3, out);

    const int iblocks = (E_DIM + 255) / 256;
    edge_idx_kernel<<<iblocks, 256, 0, stream>>>(out + ROWS);
}
